// complexNDM_26482768347376
// MI455X (gfx1250) — matrix operand order repaired and checked by reading, not compiled, not run on hardware
//
#include <hip/hip_runtime.h>
#include <hip/hip_bf16.h>

// ---------------------------------------------------------------------------
// B=64, S=256, T=256, H=1024, OUT=64, IN=128.
// ~135 GFLOP of GEMM -> v_wmma_f32_16x16x32_bf16, bf16 end-to-end between
// layers (fp32 accumulate). Global->LDS staging uses CDNA5 async DMA
// (global_load_async_to_lds_b128) with double-buffered LDS when available.
// ---------------------------------------------------------------------------

typedef __attribute__((ext_vector_type(16))) __bf16 v16bf;
typedef __attribute__((ext_vector_type(8)))  float  v8f;
typedef __attribute__((ext_vector_type(4)))  int    v4i;
typedef unsigned short ushort_t;

#define BM 128
#define BN 64
#define BK 32
#define KPAD 40   // 32+8 bf16 pad: 80B row stride (16B aligned), bank-spread

union FragBF { v16bf v; uint4 q[2]; };

#if defined(__gfx1250__) && __has_builtin(__builtin_amdgcn_global_load_async_to_lds_b128)
#define ASYNC_LDS 1
#else
#define ASYNC_LDS 0
#endif

typedef __attribute__((address_space(1))) v4i gv4i;   // global int4
typedef __attribute__((address_space(3))) v4i lv4i;   // LDS int4

// 16-byte global->LDS copy: async DMA if the toolchain has it, else sync.
__device__ __forceinline__ void cp16(ushort_t* lds_dst, const ushort_t* gsrc)
{
#if ASYNC_LDS
    __builtin_amdgcn_global_load_async_to_lds_b128(
        (gv4i*)(unsigned long long)(uintptr_t)gsrc,
        (lv4i*)(unsigned int)(uintptr_t)lds_dst,
        0, 0);
#else
    *(uint4*)lds_dst = *(const uint4*)gsrc;
#endif
}

__device__ __forceinline__ void wait_async_lds()
{
#if ASYNC_LDS
#if __has_builtin(__builtin_amdgcn_s_wait_asynccnt)
    __builtin_amdgcn_s_wait_asynccnt(0);
#else
    asm volatile("s_wait_asynccnt 0x0" ::: "memory");
#endif
#endif
}

__device__ __forceinline__ ushort_t f2bf(float x) {
    unsigned u = __float_as_uint(x);
    unsigned r = u + 0x7FFFu + ((u >> 16) & 1u);   // RNE
    return (ushort_t)(r >> 16);
}
__device__ __forceinline__ float bf2f(ushort_t x) {
    return __uint_as_float(((unsigned)x) << 16);
}
__device__ __forceinline__ float gelu_exact(float x) {
    return 0.5f * x * (1.0f + erff(x * 0.70710678118654752f));
}

// ---------------------------------------------------------------------------
// One-time fp32 -> bf16 conversion (n multiple of 4)
// ---------------------------------------------------------------------------
__global__ void __launch_bounds__(256)
cvt_f32_bf16(const float* __restrict__ s, ushort_t* __restrict__ d, int n4)
{
    int i = blockIdx.x * blockDim.x + threadIdx.x;
    if (i >= n4) return;
    float4 v = *(const float4*)(s + (size_t)i * 4);
    uint2 o;
    o.x = (unsigned)f2bf(v.x) | ((unsigned)f2bf(v.y) << 16);
    o.y = (unsigned)f2bf(v.z) | ((unsigned)f2bf(v.w) << 16);
    *(uint2*)(d + (size_t)i * 4) = o;
}

// ---------------------------------------------------------------------------
// C[M,N](bf16) = act( A[M,K](bf16) @ W[N,K](bf16)^T + bias[N](f32) )
// 256 thr = 8 waves as 4(M) x 2(N); wave = 32x32 = 2x2 wmma tiles.
// Double-buffered LDS, async DMA staging.
// ---------------------------------------------------------------------------
__global__ void __launch_bounds__(256)
gemm_bf16(const ushort_t* __restrict__ A, const ushort_t* __restrict__ W,
          const float* __restrict__ bias, ushort_t* __restrict__ C,
          int M, int N, int K, int act)
{
    __shared__ ushort_t sA[2][BM * KPAD];
    __shared__ ushort_t sB[2][BN * KPAD];

    const int tid  = threadIdx.x;
    const int lane = tid & 31;
    const int wave = tid >> 5;
    const int wm   = wave >> 1;
    const int wn   = wave & 1;
    const int blockM = blockIdx.y * BM;
    const int blockN = blockIdx.x * BN;

    const v8f vz = {0.f,0.f,0.f,0.f,0.f,0.f,0.f,0.f};
    v8f acc[2][2] = {{vz, vz}, {vz, vz}};

    const int nk = K / BK;
    const uint4 z4 = make_uint4(0u, 0u, 0u, 0u);

    auto issue = [&](int buf, int k0) {
        // A tile: 128 rows x 4 16B-chunks = 512 chunks, 2 per thread
        #pragma unroll
        for (int s = 0; s < 2; ++s) {
            int id = tid + s * 256;
            int r = id >> 2, c = (id & 3) * 8;
            ushort_t* dst = &sA[buf][r * KPAD + c];
            int gr = blockM + r;
            if (gr < M) cp16(dst, A + (size_t)gr * K + k0 + c);
            else        *(uint4*)dst = z4;
        }
        // W tile: 64 rows x 4 chunks = 256 chunks, 1 per thread
        {
            int r = tid >> 2, c = (tid & 3) * 8;
            cp16(&sB[buf][r * KPAD + c], W + (size_t)(blockN + r) * K + k0 + c);
        }
    };

    issue(0, 0);

    const int arow  = lane & 15;
    const int akoff = (lane >> 4) * 8;
    const int bkoff = (lane >> 4) * 8;

    for (int ik = 0; ik < nk; ++ik) {
        wait_async_lds();
        __syncthreads();
        if (ik + 1 < nk) issue((ik + 1) & 1, (ik + 1) * BK);

        const ushort_t* cA = sA[ik & 1];
        const ushort_t* cB = sB[ik & 1];

        FragBF a[2], b[2];
        #pragma unroll
        for (int i = 0; i < 2; ++i) {
            const ushort_t* p = &cA[(wm * 32 + i * 16 + arow) * KPAD + akoff];
            a[i].q[0] = *(const uint4*)(p);        // K koff..+7
            a[i].q[1] = *(const uint4*)(p + 16);   // K koff+16..+23
        }
        #pragma unroll
        for (int j = 0; j < 2; ++j) {
            const ushort_t* p = &cB[(wn * 32 + j * 16 + (lane & 15)) * KPAD + bkoff];
            b[j].q[0] = *(const uint4*)(p);        // K koff..+7
            b[j].q[1] = *(const uint4*)(p + 16);   // K koff+16..+23
        }
        #pragma unroll
        for (int i = 0; i < 2; ++i)
            #pragma unroll
            for (int j = 0; j < 2; ++j)
                acc[i][j] = __builtin_amdgcn_wmma_f32_16x16x32_bf16(
                    false, a[i].v, false, b[j].v, (short)0, acc[i][j], false, false);
    }

    // C/D layout: lane&15 = N; VGPR r -> M = r + 8*(lane>=16)
    const int nbase = blockN + wn * 32 + (lane & 15);
    const int mbase = blockM + wm * 32 + ((lane >> 4) * 8);
    #pragma unroll
    for (int i = 0; i < 2; ++i) {
        #pragma unroll
        for (int j = 0; j < 2; ++j) {
            int n = nbase + j * 16;
            float bv = bias[n];
            #pragma unroll
            for (int r = 0; r < 8; ++r) {
                int m = mbase + i * 16 + r;
                if (m < M) {
                    float x = acc[i][j][r] + bv;
                    if (act) x = gelu_exact(x);
                    C[(size_t)m * N + n] = f2bf(x);
                }
            }
        }
    }
}

// ---------------------------------------------------------------------------
// out[t,b,:] (f32) = hs_r[b*T+t,:] @ C_wr^T - hs_i[b*T+t,:] @ C_wi^T  (N=64)
// All operands bf16, double-buffered async staging.
// ---------------------------------------------------------------------------
__global__ void __launch_bounds__(256)
gemm_cplx_real_proj(const ushort_t* __restrict__ Ar, const ushort_t* __restrict__ Ai,
                    const ushort_t* __restrict__ Wr, const ushort_t* __restrict__ Wi,
                    float* __restrict__ Cout, int M, int K, int Tt, int Bb)
{
    __shared__ ushort_t sAr[2][BM * KPAD];
    __shared__ ushort_t sAi[2][BM * KPAD];
    __shared__ ushort_t sBr[2][BN * KPAD];
    __shared__ ushort_t sBi[2][BN * KPAD];

    const int tid  = threadIdx.x;
    const int lane = tid & 31;
    const int wave = tid >> 5;
    const int wm   = wave >> 1;
    const int wn   = wave & 1;
    const int blockM = blockIdx.y * BM;      // blockN == 0 (N==64)

    const v8f vz = {0.f,0.f,0.f,0.f,0.f,0.f,0.f,0.f};
    v8f accR[2][2] = {{vz, vz}, {vz, vz}};
    v8f accI[2][2] = {{vz, vz}, {vz, vz}};

    const int nk = K / BK;

    auto issue = [&](int buf, int k0) {
        #pragma unroll
        for (int s = 0; s < 2; ++s) {
            int id = tid + s * 256;
            int r = id >> 2, c = (id & 3) * 8;
            size_t go = (size_t)(blockM + r) * K + k0 + c;
            int lo = r * KPAD + c;
            cp16(&sAr[buf][lo], Ar + go);
            cp16(&sAi[buf][lo], Ai + go);
        }
        {
            int r = tid >> 2, c = (tid & 3) * 8;
            size_t go = (size_t)r * K + k0 + c;
            int lo = r * KPAD + c;
            cp16(&sBr[buf][lo], Wr + go);
            cp16(&sBi[buf][lo], Wi + go);
        }
    };

    issue(0, 0);

    const int arow  = lane & 15;
    const int akoff = (lane >> 4) * 8;
    const int bkoff = (lane >> 4) * 8;

    for (int ik = 0; ik < nk; ++ik) {
        wait_async_lds();
        __syncthreads();
        if (ik + 1 < nk) issue((ik + 1) & 1, (ik + 1) * BK);

        FragBF ar[2], ai[2], br[2], bi[2];
        #pragma unroll
        for (int i = 0; i < 2; ++i) {
            int ro = (wm * 32 + i * 16 + arow) * KPAD + akoff;
            ar[i].q[0] = *(const uint4*)(&sAr[ik & 1][ro]);
            ar[i].q[1] = *(const uint4*)(&sAr[ik & 1][ro + 16]);
            ai[i].q[0] = *(const uint4*)(&sAi[ik & 1][ro]);
            ai[i].q[1] = *(const uint4*)(&sAi[ik & 1][ro + 16]);
        }
        #pragma unroll
        for (int j = 0; j < 2; ++j) {
            int ro = (wn * 32 + j * 16 + (lane & 15)) * KPAD + bkoff;
            br[j].q[0] = *(const uint4*)(&sBr[ik & 1][ro]);
            br[j].q[1] = *(const uint4*)(&sBr[ik & 1][ro + 16]);
            bi[j].q[0] = *(const uint4*)(&sBi[ik & 1][ro]);
            bi[j].q[1] = *(const uint4*)(&sBi[ik & 1][ro + 16]);
        }
        #pragma unroll
        for (int i = 0; i < 2; ++i)
            #pragma unroll
            for (int j = 0; j < 2; ++j) {
                accR[i][j] = __builtin_amdgcn_wmma_f32_16x16x32_bf16(
                    false, ar[i].v, false, br[j].v, (short)0, accR[i][j], false, false);
                accI[i][j] = __builtin_amdgcn_wmma_f32_16x16x32_bf16(
                    false, ai[i].v, false, bi[j].v, (short)0, accI[i][j], false, false);
            }
    }

    const int nbase = wn * 32 + (lane & 15);
    const int mbase = blockM + wm * 32 + ((lane >> 4) * 8);
    #pragma unroll
    for (int i = 0; i < 2; ++i) {
        #pragma unroll
        for (int j = 0; j < 2; ++j) {
            int n = nbase + j * 16;
            #pragma unroll
            for (int r = 0; r < 8; ++r) {
                int m = mbase + i * 16 + r;        // m = b*T + t
                int b = m / Tt, t = m - b * Tt;
                Cout[((size_t)t * Bb + b) * 64 + n] = accR[i][j][r] - accI[i][j][r];
            }
        }
    }
}

// ---------------------------------------------------------------------------
// w_eff = [w, conj(w)],  w = exp(-exp(v_log) + i*exp(theta_log))
// ---------------------------------------------------------------------------
__global__ void make_weff(const float* __restrict__ v_log,
                          const float* __restrict__ th_log,
                          float* __restrict__ wr, float* __restrict__ wi, int Hh)
{
    int h = blockIdx.x * blockDim.x + threadIdx.x;
    if (h >= Hh) return;
    float r = expf(-expf(v_log[h]));
    float a = expf(th_log[h]);
    float s, c;
    sincosf(a, &s, &c);
    wr[h] = r * c;        wi[h] = r * s;
    wr[h + Hh] = r * c;   wi[h + Hh] = -r * s;
}

// ---------------------------------------------------------------------------
// Diagonal complex scan, one lane per (b,h): fp32 accumulate, bf16 in/out.
// Overwrites ut (bf16) with hs (bf16) in place; streams complex64 hidden
// states straight into d_out.
// ---------------------------------------------------------------------------
__global__ void __launch_bounds__(256)
lru_scan(const ushort_t* __restrict__ h0r, const ushort_t* __restrict__ h0i,
         const float* __restrict__ wr, const float* __restrict__ wi,
         ushort_t* __restrict__ utr, ushort_t* __restrict__ uti,
         float2* __restrict__ hidden, int Bc, int Tt, int H)
{
    int idx = blockIdx.x * blockDim.x + threadIdx.x;
    if (idx >= Bc * H) return;
    int b = idx / H, h = idx - b * H;
    float hr = bf2f(h0r[idx]), hi = bf2f(h0i[idx]);
    float cr = wr[h], ci = wi[h];
    hidden[(size_t)b * H + h] = make_float2(hr, hi);           // hidden[0] = h0
    size_t base = (size_t)b * Tt * H + h;
    for (int t = 0; t < Tt; ++t) {
        size_t off = base + (size_t)t * H;
        float ur = bf2f(utr[off]), ui = bf2f(uti[off]);
        float nr = hr * cr - hi * ci + ur;
        float ni = hr * ci + hi * cr + ui;
        hr = nr; hi = ni;
        utr[off] = f2bf(hr); uti[off] = f2bf(hi);              // hs for out-proj
        hidden[((size_t)(t + 1) * Bc + b) * H + h] = make_float2(hr, hi);
    }
}

// ---------------------------------------------------------------------------
extern "C" void kernel_launch(void* const* d_in, const int* in_sizes, int n_in,
                              void* d_out, int out_size, void* d_ws, size_t ws_size,
                              hipStream_t stream)
{
    (void)in_sizes; (void)n_in; (void)out_size; (void)ws_size;
    const int Bc = 64, S = 256, Tt = 256, H = 1024, OUT = 64, IN = 128;
    const int MT = Bc * Tt;                       // 16384 tokens

    const float* x0    = (const float*)d_in[0];
    const float* u     = (const float*)d_in[1];
    const float* v_log = (const float*)d_in[2];
    const float* t_log = (const float*)d_in[3];
    const float* C_wr  = (const float*)d_in[4];
    const float* C_wi  = (const float*)d_in[5];
    const float* f0_w0 = (const float*)d_in[6];
    const float* f0_b0 = (const float*)d_in[7];
    const float* f0_w1 = (const float*)d_in[8];
    const float* f0_b1 = (const float*)d_in[9];
    const float* f0_w2 = (const float*)d_in[10];
    const float* f0_b2 = (const float*)d_in[11];
    const float* f0_owr= (const float*)d_in[12];
    const float* f0_owi= (const float*)d_in[13];
    const float* f0_obr= (const float*)d_in[14];
    const float* f0_obi= (const float*)d_in[15];
    const float* fu_w0 = (const float*)d_in[16];
    const float* fu_b0 = (const float*)d_in[17];
    const float* fu_w1 = (const float*)d_in[18];
    const float* fu_b1 = (const float*)d_in[19];
    const float* fu_w2 = (const float*)d_in[20];
    const float* fu_b2 = (const float*)d_in[21];
    const float* fu_owr= (const float*)d_in[22];
    const float* fu_owi= (const float*)d_in[23];
    const float* fu_obr= (const float*)d_in[24];
    const float* fu_obi= (const float*)d_in[25];

    float*  outf   = (float*)d_out;                              // (T,B,OUT)
    float2* hidden = (float2*)(outf + (size_t)Tt * Bc * OUT);    // (T+1,B,H) c64

    // ---- bf16 workspace carve-out ----
    ushort_t* p = (ushort_t*)d_ws;
    auto take = [&](size_t n) { ushort_t* r = p; p += n; return r; };
    const size_t HH = (size_t)H * H;                // 1M
    ushort_t* ub     = take((size_t)MT * IN);       // 2M
    ushort_t* x0b    = take((size_t)Bc * S * OUT);  // 1M
    ushort_t* fuw0b  = take((size_t)H * IN);
    ushort_t* fuw1b  = take(HH);
    ushort_t* fuw2b  = take(HH);
    ushort_t* fuowrb = take(HH);
    ushort_t* fuowib = take(HH);
    ushort_t* f0w0b  = take((size_t)H * S * OUT);   // 16M
    ushort_t* f0w1b  = take(HH);
    ushort_t* f0w2b  = take(HH);
    ushort_t* f0owrb = take(HH);
    ushort_t* f0owib = take(HH);
    ushort_t* cwrb   = take((size_t)OUT * H);
    ushort_t* cwib   = take((size_t)OUT * H);
    ushort_t* abuf0  = take((size_t)MT * H);        // 16M
    ushort_t* abuf1  = take((size_t)MT * H);        // 16M
    ushort_t* utrb   = take((size_t)MT * H);        // 16M
    ushort_t* utib   = take((size_t)MT * H);        // 16M
    ushort_t* hb0    = take((size_t)Bc * H);
    ushort_t* hb1    = take((size_t)Bc * H);
    ushort_t* h0rb   = take((size_t)Bc * H);
    ushort_t* h0ib   = take((size_t)Bc * H);
    float* wr = (float*)p;
    float* wi = wr + H;

    dim3 blk(256);
    auto cgrid = [](size_t n) { return dim3((unsigned)((n / 4 + 255) / 256)); };
    auto ggrid = [](int M, int N) { return dim3((N + BN - 1) / BN, (M + BM - 1) / BM); };

    // ---- one-time fp32 -> bf16 conversions (memory-bound prepass) ----
    cvt_f32_bf16<<<cgrid((size_t)MT*IN),   blk, 0, stream>>>(u,      ub,     MT*IN/4);
    cvt_f32_bf16<<<cgrid((size_t)Bc*S*OUT),blk, 0, stream>>>(x0,     x0b,    Bc*S*OUT/4);
    cvt_f32_bf16<<<cgrid((size_t)H*IN),    blk, 0, stream>>>(fu_w0,  fuw0b,  H*IN/4);
    cvt_f32_bf16<<<cgrid(HH), blk, 0, stream>>>(fu_w1,  fuw1b,  (int)(HH/4));
    cvt_f32_bf16<<<cgrid(HH), blk, 0, stream>>>(fu_w2,  fuw2b,  (int)(HH/4));
    cvt_f32_bf16<<<cgrid(HH), blk, 0, stream>>>(fu_owr, fuowrb, (int)(HH/4));
    cvt_f32_bf16<<<cgrid(HH), blk, 0, stream>>>(fu_owi, fuowib, (int)(HH/4));
    cvt_f32_bf16<<<cgrid((size_t)H*S*OUT), blk, 0, stream>>>(f0_w0, f0w0b, H*S*OUT/4);
    cvt_f32_bf16<<<cgrid(HH), blk, 0, stream>>>(f0_w1,  f0w1b,  (int)(HH/4));
    cvt_f32_bf16<<<cgrid(HH), blk, 0, stream>>>(f0_w2,  f0w2b,  (int)(HH/4));
    cvt_f32_bf16<<<cgrid(HH), blk, 0, stream>>>(f0_owr, f0owrb, (int)(HH/4));
    cvt_f32_bf16<<<cgrid(HH), blk, 0, stream>>>(f0_owi, f0owib, (int)(HH/4));
    cvt_f32_bf16<<<cgrid((size_t)OUT*H), blk, 0, stream>>>(C_wr, cwrb, OUT*H/4);
    cvt_f32_bf16<<<cgrid((size_t)OUT*H), blk, 0, stream>>>(C_wi, cwib, OUT*H/4);

    make_weff<<<dim3(2), blk, 0, stream>>>(v_log, t_log, wr, wi, H / 2);

    // ---- f_u MLP over all tokens (bulk of the FLOPs) ----
    gemm_bf16<<<ggrid(MT, H), blk, 0, stream>>>(ub,    fuw0b,  fu_b0,  abuf0, MT, H, IN, 1);
    gemm_bf16<<<ggrid(MT, H), blk, 0, stream>>>(abuf0, fuw1b,  fu_b1,  abuf1, MT, H, H,  1);
    gemm_bf16<<<ggrid(MT, H), blk, 0, stream>>>(abuf1, fuw2b,  fu_b2,  abuf0, MT, H, H,  1);
    gemm_bf16<<<ggrid(MT, H), blk, 0, stream>>>(abuf0, fuowrb, fu_obr, utrb,  MT, H, H,  0);
    gemm_bf16<<<ggrid(MT, H), blk, 0, stream>>>(abuf0, fuowib, fu_obi, utib,  MT, H, H,  0);

    // ---- f_0 MLP (M=64, K=16384 first layer) ----
    gemm_bf16<<<ggrid(Bc, H), blk, 0, stream>>>(x0b, f0w0b,  f0_b0,  hb0,  Bc, H, S * OUT, 1);
    gemm_bf16<<<ggrid(Bc, H), blk, 0, stream>>>(hb0, f0w1b,  f0_b1,  hb1,  Bc, H, H, 1);
    gemm_bf16<<<ggrid(Bc, H), blk, 0, stream>>>(hb1, f0w2b,  f0_b2,  hb0,  Bc, H, H, 1);
    gemm_bf16<<<ggrid(Bc, H), blk, 0, stream>>>(hb0, f0owrb, f0_obr, h0rb, Bc, H, H, 0);
    gemm_bf16<<<ggrid(Bc, H), blk, 0, stream>>>(hb0, f0owib, f0_obi, h0ib, Bc, H, H, 0);

    // ---- diagonal complex recurrence (65536 lanes, serial in T) ----
    lru_scan<<<dim3((Bc * H + 255) / 256), blk, 0, stream>>>(
        h0rb, h0ib, wr, wi, utrb, utib, hidden, Bc, Tt, H);

    // ---- outputs = Re(hs @ Cw^T), permuted (b*T+t) -> (t,b) store ----
    gemm_cplx_real_proj<<<dim3(1, MT / BM), blk, 0, stream>>>(
        utrb, utib, cwrb, cwib, outf, MT, H, Tt, Bc);
}
